// MaskedTabularTransformer_58102317580700
// MI455X (gfx1250) — compile-verified, matrix operand order checked by reading, not run on hardware
//
#include <hip/hip_runtime.h>
#include <hip/hip_bf16.h>
#include <stdint.h>

// ---------------- types ----------------
typedef __attribute__((ext_vector_type(16))) __bf16 bf16x16;
typedef __attribute__((ext_vector_type(8)))  __bf16 bf16x8;
typedef __attribute__((ext_vector_type(8)))  float  f32x8;
typedef __attribute__((ext_vector_type(4)))  unsigned int u32x4;

union FragAB { bf16x16 v; bf16x8 h[2]; };

__device__ __forceinline__ unsigned short f2bf(float f) {
  union { float f; unsigned int u; } a; a.f = f;
  unsigned int r = a.u + 0x7FFFu + ((a.u >> 16) & 1u);
  return (unsigned short)(r >> 16);
}

// gfx1250 async global->LDS copy (ASYNCcnt-tracked, no VGPR round-trip).
__device__ __forceinline__ void async_load_b128(unsigned lds_off, const void* gaddr) {
  asm volatile("global_load_async_to_lds_b128 %0, %1, off"
               :: "v"(lds_off), "v"((unsigned long long)(size_t)gaddr)
               : "memory");
}
__device__ __forceinline__ void wait_async0() {
  asm volatile("s_wait_asynccnt 0x0" ::: "memory");
}

// ---------------- fp32 -> bf16 conversion ----------------
__global__ void f32_to_bf16_kernel(const float* __restrict__ src,
                                   unsigned short* __restrict__ dst, long n) {
  long i = (long)blockIdx.x * blockDim.x + threadIdx.x;
  if (i < n) dst[i] = f2bf(src[i]);
}

// ---------------- token construction ----------------
// h[b, s, d] for b<2048, s<64, d<512  (SEQ = 1 cls + 59 num + sT + tT + sY + tY)
__global__ void build_tokens_kernel(const float* __restrict__ x,    // [B,59]
                                    const int*   __restrict__ t,    // [B]
                                    const float* __restrict__ Wnum, // [59,512]
                                    const float* __restrict__ bnum, // [59,512]
                                    const float* __restrict__ tokT, // [3,512]
                                    const float* __restrict__ tokY, // [33,512]
                                    const float* __restrict__ tokS, // [3,512]
                                    const float* __restrict__ pos,  // [65,512]
                                    float* __restrict__ h,          // [B*64,512]
                                    unsigned short* __restrict__ hbf) {
  long i = (long)blockIdx.x * 256 + threadIdx.x;
  const long total = 2048L * 64 * 512;
  if (i >= total) return;
  int d = (int)(i & 511);
  long bs = i >> 9;
  int s = (int)(bs & 63);
  long b = bs >> 6;
  float val;
  if (s == 0) {
    val = tokS[d];
  } else if (s <= 59) {
    int f = s - 1;
    val = x[b * 59 + f] * Wnum[(long)f * 512 + d] + bnum[(long)f * 512 + d];
  } else if (s == 60) {
    val = tokS[512 + d];
  } else if (s == 61) {
    int tt = t[b];
    if (tt == -1) tt = 2;
    val = tokT[(long)tt * 512 + d];
  } else if (s == 62) {
    val = tokS[1024 + d];
  } else {  // s == 63 : y is NaN in forward -> mask token tokY[32]
    val = tokY[32L * 512 + d];
  }
  val += pos[(long)s * 512 + d];
  h[i] = val;
  hbf[i] = f2bf(val);
}

// ---------------- generic WMMA GEMM: C = act(A @ W^T + bias) ----------------
// A: [M, lda] bf16 (M = gridDim.y*128, no M guard), W: [N, ldw] bf16, bias fp32.
// Tile 128x128x64; 8 waves, each wave = 2(M) x 4(N) 16x16 tiles, 16 wmma/round.
// Staging via gfx1250 async global->LDS (A always; W when N%128==0).
__global__ __launch_bounds__(256) void wmma_gemm_kernel(
    const unsigned short* __restrict__ A, long lda,
    const unsigned short* __restrict__ W, long ldw,
    const float* __restrict__ bias,
    void* __restrict__ Cv, long ldc,
    int N, int K, int relu, int out_bf16) {
  __shared__ unsigned short As[128][72];  // 144B rows; 16B-aligned chunks
  __shared__ unsigned short Bs[128][72];

  const long bm = (long)blockIdx.y * 128;
  const int  bn = (int)blockIdx.x * 128;
  const int tid = threadIdx.x;
  const int lane = tid & 31;
  const int wave = tid >> 5;
  const int wm = (wave >> 1) * 32;  // 0,32,64,96
  const int wn = (wave & 1) * 64;   // 0,64
  const int lr = tid >> 3;          // 0..31
  const int lc = (tid & 7) * 8;     // 0,8,...,56

  const int am = lane & 15;
  const int khalf = (lane & 16) ? 8 : 0;
  const int mh = (lane & 16) ? 8 : 0;
  const bool nfull = ((N & 127) == 0);

  const f32x8 zacc = {0.f, 0.f, 0.f, 0.f, 0.f, 0.f, 0.f, 0.f};
  f32x8 acc[2][4];
#pragma unroll
  for (int mt = 0; mt < 2; ++mt)
#pragma unroll
    for (int nt = 0; nt < 4; ++nt) acc[mt][nt] = zacc;

  const u32x4 vz = {0u, 0u, 0u, 0u};

  for (int k0 = 0; k0 < K; k0 += 64) {
    // ---- stage A tile (async, no VGPR round-trip) ----
#pragma unroll
    for (int it = 0; it < 4; ++it) {
      int row = lr + it * 32;
      async_load_b128((unsigned)(size_t)&As[row][lc],
                      A + (size_t)(bm + row) * lda + k0 + lc);
    }
    // ---- stage W tile ----
    if (nfull) {
#pragma unroll
      for (int it = 0; it < 4; ++it) {
        int row = lr + it * 32;
        async_load_b128((unsigned)(size_t)&Bs[row][lc],
                        W + (size_t)(bn + row) * ldw + k0 + lc);
      }
    } else {
#pragma unroll
      for (int it = 0; it < 4; ++it) {
        int row = lr + it * 32;
        u32x4 vb = vz;
        if (bn + row < N) vb = *(const u32x4*)(W + (size_t)(bn + row) * ldw + k0 + lc);
        *(u32x4*)(&Bs[row][lc]) = vb;
      }
    }
    // prefetch next K tile into caches while this round computes
    if (k0 + 64 < K) {
      __builtin_prefetch(A + (size_t)(bm + lr) * lda + k0 + 64 + lc, 0, 1);
      if (nfull) __builtin_prefetch(W + (size_t)(bn + lr) * ldw + k0 + 64 + lc, 0, 1);
    }
    wait_async0();
    __syncthreads();

#pragma unroll
    for (int ks = 0; ks < 2; ++ks) {
      const int kb = ks * 32 + khalf;
      FragAB fa[2], fb[4];
#pragma unroll
      for (int mt = 0; mt < 2; ++mt) {
        const unsigned short* p = &As[wm + mt * 16 + am][kb];
        fa[mt].h[0] = *(const bf16x8*)(p);
        fa[mt].h[1] = *(const bf16x8*)(p + 16);
      }
#pragma unroll
      for (int nt = 0; nt < 4; ++nt) {
        const unsigned short* p = &Bs[wn + nt * 16 + am][kb];
        fb[nt].h[0] = *(const bf16x8*)(p);
        fb[nt].h[1] = *(const bf16x8*)(p + 16);
      }
#pragma unroll
      for (int mt = 0; mt < 2; ++mt)
#pragma unroll
        for (int nt = 0; nt < 4; ++nt)
          acc[mt][nt] = __builtin_amdgcn_wmma_f32_16x16x32_bf16(
              false, fa[mt].v, false, fb[nt].v, (short)0, acc[mt][nt], false, false);
    }
    __syncthreads();
  }

  // epilogue: C layout -> lane holds col n = am, rows m = r + (lane>=16 ? 8:0)
#pragma unroll
  for (int mt = 0; mt < 2; ++mt) {
#pragma unroll
    for (int nt = 0; nt < 4; ++nt) {
      int gn = bn + wn + nt * 16 + am;
      if (gn >= N) continue;
      float bv = bias ? bias[gn] : 0.f;
#pragma unroll
      for (int r = 0; r < 8; ++r) {
        size_t gm = (size_t)(bm + wm + mt * 16 + r + mh);
        float v = acc[mt][nt][r] + bv;
        if (relu) v = fmaxf(v, 0.f);
        if (out_bf16) ((unsigned short*)Cv)[gm * ldc + gn] = f2bf(v);
        else          ((float*)Cv)[gm * ldc + gn] = v;
      }
    }
  }
}

// ---------------- fused attention per (batch, head) ----------------
// qkv: [B*64, 1536] bf16 (cols 0..511 Q | 512..1023 K | 1024..1535 V, head h at h*64)
// out: [B*64, 512] bf16.  SEQ = HD = 64 -> 4x4 grid of 16x16 WMMA tiles.
__global__ __launch_bounds__(128) void attn_kernel(
    const unsigned short* __restrict__ qkv, unsigned short* __restrict__ out) {
  __shared__ unsigned short Vt[64][72];      // [hd][key], 144B rows
  __shared__ unsigned short Pw[4][16][72];   // per-wave softmax probs [qrow][key]

  const int blk = blockIdx.x;
  const int b = blk >> 3;
  const int hh = blk & 7;
  const size_t base = (size_t)b * 64 * 1536 + (size_t)hh * 64;
  const int tid = threadIdx.x;
  const int lane = tid & 31;
  const int wave = tid >> 5;
  const int am = lane & 15;
  const int khalf = (lane & 16) ? 8 : 0;
  const int mh = (lane & 16) ? 8 : 0;
  const int m0 = wave * 16;

  // stage V transposed into LDS (contiguous 16B global reads, scalar LDS writes)
#pragma unroll
  for (int j = 0; j < 4; ++j) {
    int vi = tid + j * 128;   // 0..511
    int key = vi >> 3;
    int hdg = (vi & 7) * 8;
    union { u32x4 v; unsigned short s[8]; } u;
    u.v = *(const u32x4*)(qkv + base + (size_t)key * 1536 + 1024 + hdg);
#pragma unroll
    for (int e = 0; e < 8; ++e) Vt[hdg + e][key] = u.s[e];
  }
  __syncthreads();

  const f32x8 zacc = {0.f, 0.f, 0.f, 0.f, 0.f, 0.f, 0.f, 0.f};

  // S = Q @ K^T  (M=16 q rows per wave, N=64 keys, K=HD=64)
  f32x8 sacc[4] = {zacc, zacc, zacc, zacc};
#pragma unroll
  for (int kk = 0; kk < 2; ++kk) {
    int kb = kk * 32 + khalf;
    FragAB fq;
    const unsigned short* qp = qkv + base + (size_t)(m0 + am) * 1536 + kb;
    fq.h[0] = *(const bf16x8*)(qp);
    fq.h[1] = *(const bf16x8*)(qp + 16);
#pragma unroll
    for (int nt = 0; nt < 4; ++nt) {
      FragAB fk;
      const unsigned short* kp = qkv + base + (size_t)(nt * 16 + am) * 1536 + 512 + kb;
      fk.h[0] = *(const bf16x8*)(kp);
      fk.h[1] = *(const bf16x8*)(kp + 16);
      sacc[nt] = __builtin_amdgcn_wmma_f32_16x16x32_bf16(
          false, fq.v, false, fk.v, (short)0, sacc[nt], false, false);
    }
  }

  // softmax over each q row (row lives across one 16-lane half + 4 N tiles)
  const float SC = 0.125f;  // 1/sqrt(64)
#pragma unroll
  for (int r = 0; r < 8; ++r) {
    float mx = -3.4e38f;
#pragma unroll
    for (int nt = 0; nt < 4; ++nt) {
      sacc[nt][r] = sacc[nt][r] * SC;
      mx = fmaxf(mx, sacc[nt][r]);
    }
#pragma unroll
    for (int m = 1; m < 16; m <<= 1) mx = fmaxf(mx, __shfl_xor(mx, m, 32));
    float sum = 0.f;
#pragma unroll
    for (int nt = 0; nt < 4; ++nt) {
      float e = __expf(sacc[nt][r] - mx);
      sacc[nt][r] = e;
      sum += e;
    }
#pragma unroll
    for (int m = 1; m < 16; m <<= 1) sum += __shfl_xor(sum, m, 32);
    float inv = 1.f / sum;
    int lrow = r + mh;
#pragma unroll
    for (int nt = 0; nt < 4; ++nt)
      Pw[wave][lrow][nt * 16 + am] = f2bf(sacc[nt][r] * inv);
  }
  __syncthreads();

  // O = P @ V  (M=16 q rows, N=64 hd, K=64 keys)
  f32x8 oacc[4] = {zacc, zacc, zacc, zacc};
#pragma unroll
  for (int kk = 0; kk < 2; ++kk) {
    int kb = kk * 32 + khalf;
    FragAB fp;
    fp.h[0] = *(const bf16x8*)(&Pw[wave][am][kb]);
    fp.h[1] = *(const bf16x8*)(&Pw[wave][am][kb + 16]);
#pragma unroll
    for (int nt = 0; nt < 4; ++nt) {
      FragAB fv;
      fv.h[0] = *(const bf16x8*)(&Vt[nt * 16 + am][kb]);
      fv.h[1] = *(const bf16x8*)(&Vt[nt * 16 + am][kb + 16]);
      oacc[nt] = __builtin_amdgcn_wmma_f32_16x16x32_bf16(
          false, fp.v, false, fv.v, (short)0, oacc[nt], false, false);
    }
  }

  const size_t obase = (size_t)b * 64 * 512 + (size_t)hh * 64;
#pragma unroll
  for (int nt = 0; nt < 4; ++nt) {
    int col = nt * 16 + am;
#pragma unroll
    for (int r = 0; r < 8; ++r) {
      size_t row = (size_t)(m0 + r + mh);
      out[obase + row * 512 + col] = f2bf(oacc[nt][r]);
    }
  }
}

// ---------------- residual add + LayerNorm (wave per row, D=512) ----------------
__global__ __launch_bounds__(256) void residual_ln_kernel(
    const float* __restrict__ res, const float* __restrict__ add,
    const float* __restrict__ g, const float* __restrict__ beta,
    float* __restrict__ hout, unsigned short* __restrict__ hbf) {
  const size_t row = (size_t)blockIdx.x * 8 + (threadIdx.x >> 5);
  const int lane = threadIdx.x & 31;
  const float* rp = res + row * 512;
  const float* ap = add + row * 512;
  float v[16];
  float s = 0.f, sq = 0.f;
#pragma unroll
  for (int i = 0; i < 16; ++i) {
    int d = i * 32 + lane;
    float xv = rp[d] + ap[d];
    v[i] = xv;
    s += xv;
    sq += xv * xv;
  }
#pragma unroll
  for (int m = 1; m < 32; m <<= 1) {
    s += __shfl_xor(s, m, 32);
    sq += __shfl_xor(sq, m, 32);
  }
  const float mean = s * (1.f / 512.f);
  const float var = sq * (1.f / 512.f) - mean * mean;
  const float inv = rsqrtf(var + 1e-5f);
#pragma unroll
  for (int i = 0; i < 16; ++i) {
    int d = i * 32 + lane;
    float y = (v[i] - mean) * inv * g[d] + beta[d];
    hout[row * 512 + d] = y;
    hbf[row * 512 + d] = f2bf(y);
  }
}

// ---------------- orchestration ----------------
extern "C" void kernel_launch(void* const* d_in, const int* in_sizes, int n_in,
                              void* d_out, int out_size, void* d_ws, size_t ws_size,
                              hipStream_t stream) {
  (void)in_sizes; (void)n_in; (void)out_size; (void)ws_size;
  const float* x    = (const float*)d_in[0];
  const int*   t    = (const int*)d_in[1];
  const float* Wnum = (const float*)d_in[2];
  const float* bnum = (const float*)d_in[3];
  const float* tokT = (const float*)d_in[4];
  const float* tokY = (const float*)d_in[5];
  const float* tokS = (const float*)d_in[6];
  const float* pos  = (const float*)d_in[7];
  const float* Wqkv = (const float*)d_in[8];
  const float* bqkv = (const float*)d_in[9];
  const float* Wo   = (const float*)d_in[10];
  const float* bo   = (const float*)d_in[11];
  const float* g1   = (const float*)d_in[12];
  const float* be1  = (const float*)d_in[13];
  const float* W1   = (const float*)d_in[14];
  const float* b1   = (const float*)d_in[15];
  const float* W2   = (const float*)d_in[16];
  const float* b2   = (const float*)d_in[17];
  const float* g2   = (const float*)d_in[18];
  const float* be2  = (const float*)d_in[19];
  const float* WY   = (const float*)d_in[20];
  const float* bY   = (const float*)d_in[21];

  const long BS = 2048L * 64;  // rows of the token stream
  size_t off = 0;
  auto carve = [&](size_t bytes) -> void* {
    void* p = (char*)d_ws + off;
    off += (bytes + 255) & ~(size_t)255;
    return p;
  };
  float*          h     = (float*)carve((size_t)BS * 512 * 4);
  float*          tmpC  = (float*)carve((size_t)BS * 512 * 4);
  unsigned short* bufA  = (unsigned short*)carve((size_t)BS * 2048 * 2);  // qkv / ff1
  unsigned short* bufB  = (unsigned short*)carve((size_t)BS * 512 * 2);   // attn out
  unsigned short* bufD  = (unsigned short*)carve((size_t)BS * 512 * 2);   // h in bf16
  unsigned short* wqkvb = (unsigned short*)carve(6UL * 1536 * 512 * 2);
  unsigned short* wob   = (unsigned short*)carve(6UL * 512 * 512 * 2);
  unsigned short* w1b   = (unsigned short*)carve(6UL * 2048 * 512 * 2);
  unsigned short* w2b   = (unsigned short*)carve(6UL * 512 * 2048 * 2);
  unsigned short* wyb   = (unsigned short*)carve(32UL * 512 * 2);

  auto cvt = [&](const float* s, unsigned short* dptr, long n) {
    f32_to_bf16_kernel<<<(unsigned)((n + 255) / 256), 256, 0, stream>>>(s, dptr, n);
  };
  cvt(Wqkv, wqkvb, 6L * 1536 * 512);
  cvt(Wo,   wob,   6L * 512 * 512);
  cvt(W1,   w1b,   6L * 2048 * 512);
  cvt(W2,   w2b,   6L * 512 * 2048);
  cvt(WY,   wyb,   32L * 512);

  build_tokens_kernel<<<(unsigned)((BS * 512 + 255) / 256), 256, 0, stream>>>(
      x, t, Wnum, bnum, tokT, tokY, tokS, pos, h, bufD);

  const unsigned gy = (unsigned)(BS / 128);  // 1024
  for (int l = 0; l < 6; ++l) {
    // qkv = h @ Wqkv^T + bqkv   -> bufA bf16 [BS, 1536]
    wmma_gemm_kernel<<<dim3(1536 / 128, gy), 256, 0, stream>>>(
        bufD, 512, wqkvb + (size_t)l * 1536 * 512, 512, bqkv + (size_t)l * 1536,
        bufA, 1536, 1536, 512, 0, 1);
    // attention -> bufB bf16 [BS, 512]
    attn_kernel<<<2048 * 8, 128, 0, stream>>>(bufA, bufB);
    // o-proj -> tmpC fp32
    wmma_gemm_kernel<<<dim3(512 / 128, gy), 256, 0, stream>>>(
        bufB, 512, wob + (size_t)l * 512 * 512, 512, bo + (size_t)l * 512,
        tmpC, 512, 512, 512, 0, 0);
    // h = LN(h + tmpC)
    residual_ln_kernel<<<(unsigned)(BS / 8), 256, 0, stream>>>(
        h, tmpC, g1 + (size_t)l * 512, be1 + (size_t)l * 512, h, bufD);
    // ff1 = relu(h @ W1^T + b1) -> bufA bf16 [BS, 2048]
    wmma_gemm_kernel<<<dim3(2048 / 128, gy), 256, 0, stream>>>(
        bufD, 512, w1b + (size_t)l * 2048 * 512, 512, b1 + (size_t)l * 2048,
        bufA, 2048, 2048, 512, 1, 1);
    // ff2 = ff1 @ W2^T + b2 -> tmpC fp32
    wmma_gemm_kernel<<<dim3(512 / 128, gy), 256, 0, stream>>>(
        bufA, 2048, w2b + (size_t)l * 512 * 2048, 2048, b2 + (size_t)l * 512,
        tmpC, 512, 512, 2048, 0, 0);
    // h = LN(h + tmpC)
    residual_ln_kernel<<<(unsigned)(BS / 8), 256, 0, stream>>>(
        h, tmpC, g2 + (size_t)l * 512, be2 + (size_t)l * 512, h, bufD);
  }

  // head: out[b, :] = h[b, 63, :] @ WY^T + bY   (M=2048, N=32, K=512)
  wmma_gemm_kernel<<<dim3(1, 2048 / 128), 256, 0, stream>>>(
      bufD + (size_t)63 * 512, 64 * 512, wyb, 512, bY,
      d_out, 32, 32, 512, 0, 0);
}
